// MistralAudioCodebook_3642132267289
// MI455X (gfx1250) — matrix operand order repaired and checked by reading, not compiled, not run on hardware
//
#include <hip/hip_runtime.h>
#include <hip/hip_bf16.h>

typedef __attribute__((ext_vector_type(16))) _Float16 v16h;
typedef __attribute__((ext_vector_type(8)))  _Float16 v8h;
typedef __attribute__((ext_vector_type(8)))  float    v8f;

#define SEM_DIM   512
#define SEM_SIZE  2048
#define TT        4096
#define BB        16
#define CH        544      // SEM_DIM + ACO_DIM
#define ACO_DIM   32
#define CODE_CH   33       // 1 sem + 32 aco code channels
#define MT        128      // tokens per workgroup
#define NB        64       // codebook entries per wave iteration
#define LDS_PITCH 520      // 512 + 8 halves pad (16B-aligned rows, avoids bank conflicts)
#define EPSV      1e-5f
#define HALFV     4.0f

union V16 { v16h v; v8h h[2]; };

__device__ __forceinline__ int lane_id() {
  return (int)__builtin_amdgcn_mbcnt_lo(~0u, 0u);   // wave32: full lane id
}
__device__ __forceinline__ float xor_shfl_f(float v, int mask) {
  int i = __builtin_amdgcn_ds_bpermute((lane_id() ^ mask) << 2,
                                       __builtin_bit_cast(int, v));
  return __builtin_bit_cast(float, i);
}
__device__ __forceinline__ int xor_shfl_i(int v, int mask) {
  return __builtin_amdgcn_ds_bpermute((lane_id() ^ mask) << 2, v);
}

// ---------------------------------------------------------------------------
// Kernel 1: emb = embedding_sum / clip(usage, eps); f16 hi/lo split + ||e||^2
// one wave per codebook row
// ---------------------------------------------------------------------------
__global__ __launch_bounds__(256)
void prep_kernel(const float* __restrict__ embedding_sum,
                 const float* __restrict__ cluster_usage,
                 _Float16* __restrict__ emb_hi,
                 _Float16* __restrict__ emb_lo,
                 float* __restrict__ enorm) {
  const int wave = threadIdx.x >> 5;
  const int lane = threadIdx.x & 31;
  const int n = blockIdx.x * 8 + wave;
  const float u = fmaxf(cluster_usage[n], EPSV);
  float acc = 0.f;
  #pragma unroll
  for (int j = 0; j < SEM_DIM / 32; ++j) {
    const int k = j * 32 + lane;
    const float e = embedding_sum[(size_t)n * SEM_DIM + k] / u;
    const _Float16 h = (_Float16)e;
    const _Float16 l = (_Float16)(e - (float)h);
    emb_hi[(size_t)n * SEM_DIM + k] = h;
    emb_lo[(size_t)n * SEM_DIM + k] = l;
    acc += e * e;
  }
  #pragma unroll
  for (int m = 16; m >= 1; m >>= 1) acc += xor_shfl_f(acc, m);
  if (lane == 0) enorm[n] = acc;
}

// ---------------------------------------------------------------------------
// Kernel 2: distance GEMM (f16 WMMA, hi/lo 3-term) + argmin + STE select
// 512 WGs x 256 threads; each WG owns 128 tokens (fixed b), all 2048 codes
// ---------------------------------------------------------------------------
__global__ __launch_bounds__(256)
void vq_kernel(const float* __restrict__ x,
               const float* __restrict__ embedding_sum,
               const float* __restrict__ cluster_usage,
               const float* __restrict__ probs_sem,
               const _Float16* __restrict__ emb_hi,
               const _Float16* __restrict__ emb_lo,
               const float* __restrict__ enorm,
               float* __restrict__ quant_out,
               int* __restrict__ codes_out) {
  extern __shared__ _Float16 smem[];          // A_hi[128][520], A_lo[128][520]
  __shared__ int lds_codes[MT];
  _Float16* Ah = smem;
  _Float16* Al = smem + (size_t)MT * LDS_PITCH;

  const int tid  = threadIdx.x;
  const int wave = tid >> 5;
  const int lane = tid & 31;
  const int token0 = blockIdx.x * MT;         // flat (b,t) base
  const int b  = token0 / TT;
  const int t0 = token0 % TT;

  // ---- Stage A: x[b, 0:512, t0:t0+128] -> LDS f16 hi/lo, layout [token][k]
  #pragma unroll 1
  for (int it = 0; it < (SEM_DIM * MT / 4) / 256; ++it) {   // 64 iters
    const int idx = it * 256 + tid;                         // over [512][32] float4
    const int d  = idx >> 5;
    const int c4 = (idx & 31) << 2;
    const float4 v = *(const float4*)(x + (size_t)(b * CH + d) * TT + t0 + c4);
    const float vv[4] = {v.x, v.y, v.z, v.w};
    #pragma unroll
    for (int j = 0; j < 4; ++j) {
      const _Float16 h = (_Float16)vv[j];
      const _Float16 l = (_Float16)(vv[j] - (float)h);
      Ah[(size_t)(c4 + j) * LDS_PITCH + d] = h;
      Al[(size_t)(c4 + j) * LDS_PITCH + d] = l;
    }
  }
  __syncthreads();

  // ---- GEMM + running per-lane argmin -------------------------------------
  const int half16 = lane >> 4;               // 0 or 1
  const int l16    = lane & 15;
  const _Float16* aBaseH = Ah + (size_t)(wave * 16 + l16) * LDS_PITCH + half16 * 16;
  const _Float16* aBaseL = Al + (size_t)(wave * 16 + l16) * LDS_PITCH + half16 * 16;

  float rmin[8];
  int   ridx[8];
  #pragma unroll
  for (int v = 0; v < 8; ++v) { rmin[v] = 3.4e38f; ridx[v] = 0; }

  const v8f vzero = {0.f, 0.f, 0.f, 0.f, 0.f, 0.f, 0.f, 0.f};

  for (int n0 = 0; n0 < SEM_SIZE; n0 += NB) {
    v8f c[4];
    #pragma unroll
    for (int nn = 0; nn < 4; ++nn) c[nn] = vzero;

    #pragma unroll 2
    for (int k0 = 0; k0 < SEM_DIM; k0 += 32) {
      V16 ah, al;
      ah.h[0] = *(const v8h*)(aBaseH + k0);
      ah.h[1] = *(const v8h*)(aBaseH + k0 + 8);
      al.h[0] = *(const v8h*)(aBaseL + k0);
      al.h[1] = *(const v8h*)(aBaseL + k0 + 8);
      #pragma unroll
      for (int nn = 0; nn < 4; ++nn) {
        const int n = n0 + nn * 16 + l16;
        const size_t boff = (size_t)n * SEM_DIM + k0 + half16 * 16;
        const v16h vbh = *(const v16h*)(emb_hi + boff);
        const v16h vbl = *(const v16h*)(emb_lo + boff);
        c[nn] = __builtin_amdgcn_wmma_f32_16x16x32_f16(
            false, ah.v, false, vbh, (short)0, c[nn], false, false);
        c[nn] = __builtin_amdgcn_wmma_f32_16x16x32_f16(
            false, ah.v, false, vbl, (short)0, c[nn], false, false);
        c[nn] = __builtin_amdgcn_wmma_f32_16x16x32_f16(
            false, al.v, false, vbh, (short)0, c[nn], false, false);
      }
    }
    // distances d2 = ||e||^2 - 2*x.e (row-constant ||x||^2 dropped)
    #pragma unroll
    for (int nn = 0; nn < 4; ++nn) {
      const int n = n0 + nn * 16 + l16;
      const float en = enorm[n];
      #pragma unroll
      for (int v = 0; v < 8; ++v) {
        const float d2 = en - 2.0f * c[nn][v];
        if (d2 < rmin[v]) { rmin[v] = d2; ridx[v] = n; }   // n increases => first-min tie-break
      }
    }
  }

  // ---- cross-lane argmin (16-lane butterfly per half), write codes --------
  #pragma unroll
  for (int v = 0; v < 8; ++v) {
    float mv = rmin[v];
    int   mi = ridx[v];
    #pragma unroll
    for (int m = 8; m >= 1; m >>= 1) {
      const float ov = xor_shfl_f(mv, m);
      const int   oi = xor_shfl_i(mi, m);
      if (ov < mv || (ov == mv && oi < mi)) { mv = ov; mi = oi; }
    }
    if (l16 == 0) {
      const int row = v + 8 * half16;         // C layout: M = v + 8*(lane/16)
      const int tok = wave * 16 + row;
      lds_codes[tok] = mi;
      codes_out[(size_t)(b * CODE_CH) * TT + t0 + tok] = mi;
    }
  }
  __syncthreads();

  // ---- STE forward: sem_out = sel ? emb[code] : x -------------------------
  const bool sel = probs_sem[b] < 0.5f;
  #pragma unroll 1
  for (int it = 0; it < (SEM_DIM * MT) / 256; ++it) {       // 256 iters
    const int idx = it * 256 + tid;
    const int d   = idx >> 7;
    const int tok = idx & 127;
    const size_t off = (size_t)(b * CH + d) * TT + t0 + tok;
    float outv;
    if (sel) {
      const int cc = lds_codes[tok];
      const float u = fmaxf(cluster_usage[cc], EPSV);
      outv = embedding_sum[(size_t)cc * SEM_DIM + d] / u;   // exact fp32 emb row
    } else {
      outv = x[off];
    }
    quant_out[off] = outv;
  }
}

// ---------------------------------------------------------------------------
// Kernel 3: ACO path (elementwise): tanh -> quant / dither / pass, codes
// ---------------------------------------------------------------------------
__global__ __launch_bounds__(256)
void aco_kernel(const float* __restrict__ x,
                const float* __restrict__ noise,
                const float* __restrict__ probs_aco,
                float* __restrict__ quant_out,
                int* __restrict__ codes_out) {
  const int i = blockIdx.x * 256 + threadIdx.x;     // over B*ACO_DIM*T
  const int t = i & (TT - 1);
  const int d = (i >> 12) & (ACO_DIM - 1);
  const int b = i >> 17;
  const float a  = x[(size_t)(b * CH + SEM_DIM + d) * TT + t];
  const float zb = tanhf(a) * HALFV;
  const float p  = probs_aco[b];
  const float zq = rintf(zb);                       // fwd of STE round (RNE)
  const float ns = (noise[i] * 2.0f - 1.0f) * (HALFV / 9.0f);
  const float zd = fminf(fmaxf(zb + ns, -HALFV), HALFV);
  const float zo = (p < 0.5f) ? zq : ((p < 0.75f) ? zd : zb);
  const int code = (int)fminf(fmaxf(rintf(zo + HALFV), 0.0f), 8.0f);
  quant_out[(size_t)(b * CH + SEM_DIM + d) * TT + t] = zo * 0.25f;
  codes_out[(size_t)(b * CODE_CH + 1 + d) * TT + t] = code;
}

// ---------------------------------------------------------------------------
extern "C" void kernel_launch(void* const* d_in, const int* in_sizes, int n_in,
                              void* d_out, int out_size, void* d_ws, size_t ws_size,
                              hipStream_t stream) {
  const float* x             = (const float*)d_in[0];
  const float* embedding_sum = (const float*)d_in[1];
  const float* cluster_usage = (const float*)d_in[2];
  const float* noise         = (const float*)d_in[3];
  const float* probs_sem     = (const float*)d_in[4];
  const float* probs_aco     = (const float*)d_in[5];

  // workspace: emb_hi (2MB) | emb_lo (2MB) | enorm (8KB)
  _Float16* emb_hi = (_Float16*)d_ws;
  _Float16* emb_lo = emb_hi + (size_t)SEM_SIZE * SEM_DIM;
  float*    enorm  = (float*)(emb_lo + (size_t)SEM_SIZE * SEM_DIM);

  float* quant = (float*)d_out;
  int*   codes = (int*)(quant + (size_t)BB * CH * TT);

  prep_kernel<<<SEM_SIZE / 8, 256, 0, stream>>>(embedding_sum, cluster_usage,
                                                emb_hi, emb_lo, enorm);

  const size_t smem_bytes = (size_t)2 * MT * LDS_PITCH * sizeof(_Float16); // 266,240 B
  vq_kernel<<<(BB * TT) / MT, 256, smem_bytes, stream>>>(
      x, embedding_sum, cluster_usage, probs_sem,
      emb_hi, emb_lo, enorm, quant, codes);

  aco_kernel<<<(BB * ACO_DIM * TT) / 256, 256, 0, stream>>>(
      x, noise, probs_aco, quant, codes);
}
